// PheromoneRouter_34084860461137
// MI455X (gfx1250) — matrix operand order repaired and checked by reading, not compiled, not run on hardware
//
#include <hip/hip_runtime.h>
#include <hip/hip_bf16.h>

// ---------------------------------------------------------------------------
// Types for WMMA fragments (wave32, gfx1250)
// ---------------------------------------------------------------------------
typedef __attribute__((ext_vector_type(16))) __bf16         bf16x16;
typedef __attribute__((ext_vector_type(8)))  float          f32x8;
typedef __attribute__((ext_vector_type(8)))  unsigned short u16x8;
typedef __attribute__((ext_vector_type(16))) unsigned short u16x16;

union Frag { u16x16 u; bf16x16 b; };

__device__ __forceinline__ unsigned short f2bf(float f) {
  union { float f; unsigned u; } c; c.f = f;
  unsigned u = c.u;
  u += 0x7FFFu + ((u >> 16) & 1u);   // round-to-nearest-even
  return (unsigned short)(u >> 16);
}
__device__ __forceinline__ float bf2f(unsigned short h) {
  union { unsigned u; float f; } c; c.u = ((unsigned)h) << 16;
  return c.f;
}

// Async global->LDS copy of 16 bytes (gfx1250, tracked by ASYNCcnt).
// Inline asm keeps it portable across ROCm / upstream toolchain builtin skews.
__device__ __forceinline__ void async_copy_b128(unsigned lds_off, const void* gaddr) {
  asm volatile("global_load_async_to_lds_b128 %0, %1, off"
               :: "v"(lds_off), "v"(gaddr) : "memory");
}
__device__ __forceinline__ void wait_async_all() {
  asm volatile("s_wait_asynccnt 0" ::: "memory");
}

// ---------------------------------------------------------------------------
// Elementwise converts
// ---------------------------------------------------------------------------
__global__ void cvt_f32_bf16(const float* __restrict__ src,
                             unsigned short* __restrict__ dst, long n) {
  long i = (long)blockIdx.x * blockDim.x + threadIdx.x;
  if (i < n) dst[i] = f2bf(src[i]);
}

// WT[n*Kd + k] = bf16( W[(rowOff + k)*Nc + n] )   (transpose + convert)
__global__ void transpose_cvt(const float* __restrict__ W,
                              unsigned short* __restrict__ WT,
                              int Kd, int Nc, int rowOff) {
  int i = blockIdx.x * blockDim.x + threadIdx.x;
  if (i >= Kd * Nc) return;
  int n = i / Kd, k = i - n * Kd;
  WT[i] = f2bf(W[(size_t)(rowOff + k) * Nc + n]);
}

// ---------------------------------------------------------------------------
// Tiled bf16 GEMM:  C[M x N] = A[M x K] @ BT[N x K]^T (+ bias)
//   - 8-wave (256 thread) workgroup computes a 128x64 C macro-tile
//   - K-step 32: A(128x32) + B(64x32) bf16 tiles staged in LDS via
//     global_load_async_to_lds_b128 (ASYNCcnt), published with barrier
//   - each wave: 32x32 output = 2x2 v_wmma_f32_16x16x32_bf16 with fragment
//     reuse (4 WMMA per 8 ds_load_b128)
// mode: 0 = store f32, 1 = store bf16, 2 = accumulate into existing f32 C.
// Requires M % 128 == 0, N % 64 == 0, K % 32 == 0.
// ---------------------------------------------------------------------------
#define GT_M 128
#define GT_N 64
#define GT_K 32

__global__ void __launch_bounds__(256)
gemm_bf16(const unsigned short* __restrict__ A,
          const unsigned short* __restrict__ BT,
          const float* __restrict__ bias,
          void* __restrict__ C,
          int M, int N, int K, int mode)
{
  __shared__ unsigned short Asm[GT_M * GT_K];   // 8 KB
  __shared__ unsigned short Bsm[GT_N * GT_K];   // 4 KB

  const int tid  = threadIdx.x;
  const int lane = tid & 31;
  const int wv   = tid >> 5;            // 0..7
  const int wm   = wv & 3;              // wave's M quarter (4 x 32 = 128)
  const int wn   = wv >> 2;             // wave's N half    (2 x 32 = 64)
  const int l16  = lane & 15, hi = lane >> 4;

  const int tilesN  = N / GT_N;
  const int bm      = blockIdx.x / tilesN;
  const int bn      = blockIdx.x - bm * tilesN;
  const int rowBase = bm * GT_M;
  const int colBase = bn * GT_N;

  const unsigned aBase = (unsigned)(size_t)(void*)Asm;
  const unsigned bBase = (unsigned)(size_t)(void*)Bsm;

  // 16B-chunk assignments: A tile = 512 chunks (2/thread), B tile = 256 (1/thread)
  const int ra0 = tid >> 2,          pa0 = tid & 3;
  const int ra1 = (tid + 256) >> 2,  pa1 = tid & 3;   // (tid+256)&3 == tid&3
  const int rb0 = tid >> 2,          pb0 = tid & 3;

  f32x8 acc[2][2] = {};

  for (int k0 = 0; k0 < K; k0 += GT_K) {
    // ---- stage global -> LDS (async, per-lane 16B) ----
    async_copy_b128(aBase + (unsigned)(ra0 * GT_K + pa0 * 8) * 2,
                    A + (size_t)(rowBase + ra0) * K + k0 + pa0 * 8);
    async_copy_b128(aBase + (unsigned)(ra1 * GT_K + pa1 * 8) * 2,
                    A + (size_t)(rowBase + ra1) * K + k0 + pa1 * 8);
    async_copy_b128(bBase + (unsigned)(rb0 * GT_K + pb0 * 8) * 2,
                    BT + (size_t)(colBase + rb0) * K + k0 + pb0 * 8);
    wait_async_all();        // this wave's async writes are in LDS
    __syncthreads();         // publish across all 8 waves

    // ---- fragments from LDS + 2x2 WMMA ----
    Frag af[2], bf[2];
#pragma unroll
    for (int i = 0; i < 2; ++i) {
      // A 16x32 bf16 frag: lanes 0-15 K{0..7,16..23}, lanes 16-31 K{8..15,24..31}
      const unsigned short* ar = Asm + (wm * 32 + i * 16 + l16) * GT_K;
      af[i].u.lo = *(const u16x8*)(ar + hi * 8);
      af[i].u.hi = *(const u16x8*)(ar + 16 + hi * 8);
      // B 32x16 bf16 frag: lanes 0-15 K{0..7,16..23}, lanes 16-31 K{8..15,24..31}
      const unsigned short* br = Bsm + (wn * 32 + i * 16 + l16) * GT_K;
      bf[i].u.lo = *(const u16x8*)(br + hi * 8);
      bf[i].u.hi = *(const u16x8*)(br + 16 + hi * 8);
    }
#pragma unroll
    for (int i = 0; i < 2; ++i)
#pragma unroll
      for (int j = 0; j < 2; ++j)
        acc[i][j] = __builtin_amdgcn_wmma_f32_16x16x32_bf16(
            false, af[i].b, false, bf[j].b, (short)0, acc[i][j], false, false);

    __syncthreads();         // all waves done reading before next stage
  }

  // ---- epilogue ----
#pragma unroll
  for (int i = 0; i < 2; ++i) {
    const int rbase = rowBase + wm * 32 + i * 16 + hi * 8;
#pragma unroll
    for (int j = 0; j < 2; ++j) {
      const int col = colBase + wn * 32 + j * 16 + l16;
      const float bvv = bias ? bias[col] : 0.0f;
      if (mode == 1) {
        unsigned short* Cb = (unsigned short*)C;
        for (int g = 0; g < 8; ++g)
          Cb[(size_t)(rbase + g) * N + col] = f2bf(acc[i][j][g] + bvv);
      } else if (mode == 2) {
        float* Cf = (float*)C;
        for (int g = 0; g < 8; ++g)
          Cf[(size_t)(rbase + g) * N + col] += acc[i][j][g] + bvv;
      } else {
        float* Cf = (float*)C;
        for (int g = 0; g < 8; ++g)
          Cf[(size_t)(rbase + g) * N + col] = acc[i][j][g] + bvv;
      }
    }
  }
}

// ---------------------------------------------------------------------------
// s[n] = 0.25 * sum_{w=1..4} silu( Ap[n] + (n-w>=0 ? Bp[n-w] : 0) )
// Ap already contains +bm1.  Output bf16 for the following GEMM.
// ---------------------------------------------------------------------------
__global__ void silu_mean(const float* __restrict__ Ap,
                          const float* __restrict__ Bp,
                          unsigned short* __restrict__ S,
                          int Nseq, int Dd, long total)
{
  long idx = (long)blockIdx.x * blockDim.x + threadIdx.x;
  if (idx >= total) return;
  long row = idx / Dd;
  int  n   = (int)(row % Nseq);
  float a = Ap[idx];
  float s = 0.0f;
#pragma unroll
  for (int w = 1; w <= 4; ++w) {
    float pre = a;
    if (n - w >= 0) pre += Bp[idx - (long)w * Dd];
    s += pre / (1.0f + __expf(-pre));       // silu
  }
  S[idx] = f2bf(0.25f * s);
}

// ---------------------------------------------------------------------------
// Causal top-8 attention with pheromone bias.
// Block = 512 threads (16 waves) per (batch, 16-row query block).
// Phase A: waves compute 16x16 score tiles via bf16 WMMA into LDS (f32).
// Phase B: one wave per row: 8 rounds of wave-wide argmax (shfl_xor butterfly),
//          LDS masking with explicit s_wait_dscnt, softmax over 8, then the
//          8-way weighted gather of V.  Output written as bf16.
// ---------------------------------------------------------------------------
__global__ void attn_topk(const unsigned short* __restrict__ Qb,
                          const unsigned short* __restrict__ Kb,
                          const unsigned short* __restrict__ Vb,
                          const float* __restrict__ pher,
                          unsigned short* __restrict__ Gb,
                          int Nseq, int Dd)
{
  extern __shared__ float sc[];            // 16 rows x Nseq f32 scores
  const int rbCount = Nseq >> 4;
  const int b    = blockIdx.x / rbCount;
  const int rb   = blockIdx.x % rbCount;
  const int wave = threadIdx.x >> 5;
  const int lane = threadIdx.x & 31;
  const int l16  = lane & 15, hi = lane >> 4;
  const int r0   = rb << 4;
  const int mtiles = rb + 1;               // causal: m-tiles 0..rb
  const float scale = 0.03125f;            // 1/sqrt(1024)

  const unsigned short* qrow = Qb + (size_t)(b * Nseq + r0 + l16) * Dd;

  for (int mt = wave; mt < mtiles; mt += 16) {
    const int mcol = (mt << 4) + l16;
    const unsigned short* krow = Kb + (size_t)(b * Nseq + mcol) * Dd;
    f32x8 acc = {};
    for (int k0 = 0; k0 < Dd; k0 += 32) {
      Frag a, kb;
      a.u.lo = *(const u16x8*)(qrow + k0 + hi * 8);
      a.u.hi = *(const u16x8*)(qrow + k0 + 16 + hi * 8);
      const unsigned short* bp = krow + k0 + hi * 8;
      kb.u.lo = *(const u16x8*)(bp);
      kb.u.hi = *(const u16x8*)(bp + 16);
      acc = __builtin_amdgcn_wmma_f32_16x16x32_bf16(false, a.b, false, kb.b,
                                                    (short)0, acc, false, false);
    }
    const float pbias = 0.3f * pher[(size_t)b * Nseq + mcol];
    for (int g = 0; g < 8; ++g) {
      const int rloc = hi * 8 + g;
      float s = acc[g] * scale + pbias;
      if (mcol > r0 + rloc) s = -1e30f;    // causal mask
      sc[rloc * Nseq + mcol] = s;
    }
  }
  __syncthreads();

  // ---- phase B: one wave per query row ----
  const int   rloc = wave;
  float*      srow = sc + rloc * Nseq;
  const int   mlim = mtiles << 4;
  float kv[8]; int ki[8];
  for (int j = 0; j < 8; ++j) {
    float best = -3e30f; int bidx = 0;
    for (int m = lane; m < mlim; m += 32) {
      float v = srow[m];
      if (v > best) { best = v; bidx = m; }
    }
    for (int off = 16; off; off >>= 1) {
      float ov = __shfl_xor(best, off, 32);
      int   oi = __shfl_xor(bidx, off, 32);
      if (ov > best || (ov == best && oi < bidx)) { best = ov; bidx = oi; }
    }
    kv[j] = best; ki[j] = bidx;
    if (lane == 0) srow[bidx] = -1e30f;    // remove selected entry
    asm volatile("s_wait_dscnt 0" ::: "memory");
  }

  // softmax over the 8 kept scores (masked entries contribute exp(-huge)=0)
  const float mx = kv[0];
  float p[8], sum = 0.0f;
  for (int j = 0; j < 8; ++j) { p[j] = __expf(kv[j] - mx); sum += p[j]; }
  const float inv = 1.0f / sum;
  for (int j = 0; j < 8; ++j) p[j] *= inv;

  const int rglob = r0 + rloc;
  for (int d = lane; d < Dd; d += 32) {
    float a = 0.0f;
    for (int j = 0; j < 8; ++j)
      a += p[j] * bf2f(Vb[(size_t)(b * Nseq + ki[j]) * Dd + d]);
    Gb[(size_t)(b * Nseq + rglob) * Dd + d] = f2bf(a);
  }
}

// ---------------------------------------------------------------------------
// Orchestration
// ---------------------------------------------------------------------------
extern "C" void kernel_launch(void* const* d_in, const int* in_sizes, int n_in,
                              void* d_out, int out_size, void* d_ws, size_t ws_size,
                              hipStream_t stream)
{
  (void)in_sizes; (void)n_in; (void)out_size; (void)ws_size;
  const int Bb = 4, Nseq = 2048, Dd = 1024;
  const int BN = Bb * Nseq;                       // 8192

  const float* mu   = (const float*)d_in[0];
  const float* pher = (const float*)d_in[1];
  const float* Wq   = (const float*)d_in[2];
  const float* bq   = (const float*)d_in[3];
  const float* Wk   = (const float*)d_in[4];
  const float* bk   = (const float*)d_in[5];
  const float* Wv   = (const float*)d_in[6];
  const float* bv   = (const float*)d_in[7];
  const float* Wm1  = (const float*)d_in[8];
  const float* bm1  = (const float*)d_in[9];
  const float* Wm2  = (const float*)d_in[10];
  const float* bm2  = (const float*)d_in[11];
  const float* Wo   = (const float*)d_in[12];
  const float* bo   = (const float*)d_in[13];
  float* out = (float*)d_out;

  char* ws = (char*)d_ws;
  const size_t MB = 1024 * 1024;
  unsigned short* mu_bf = (unsigned short*)(ws +   0 * MB); // 16 MB
  unsigned short* WqT   = (unsigned short*)(ws +  16 * MB); //  2 MB each
  unsigned short* WkT   = (unsigned short*)(ws +  18 * MB);
  unsigned short* WvT   = (unsigned short*)(ws +  20 * MB);
  unsigned short* Wm1aT = (unsigned short*)(ws +  22 * MB);
  unsigned short* Wm1bT = (unsigned short*)(ws +  24 * MB);
  unsigned short* Wm2T  = (unsigned short*)(ws +  26 * MB);
  unsigned short* WoLT  = (unsigned short*)(ws +  28 * MB);
  unsigned short* WoGT  = (unsigned short*)(ws +  30 * MB);
  unsigned short* q_bf  = (unsigned short*)(ws +  32 * MB); // 16 MB
  unsigned short* k_bf  = (unsigned short*)(ws +  48 * MB); // 16 MB
  unsigned short* v_bf  = (unsigned short*)(ws +  64 * MB); // 16 MB
  float*          Ap    = (float*)         (ws +  80 * MB); // 32 MB
  float*          Bp    = (float*)         (ws + 112 * MB); // 32 MB
  unsigned short* s_bf  = (unsigned short*)(ws + 144 * MB); // 16 MB
  unsigned short* l_bf  = (unsigned short*)(ws + 160 * MB); // 16 MB
  unsigned short* g_bf  = (unsigned short*)(ws + 176 * MB); // 16 MB  (end 192 MB)

  const long totalMu  = (long)BN * Dd;                       // 8,388,608
  const int  eltBlk   = 256;
  const int  muBlocks = (int)((totalMu + eltBlk - 1) / eltBlk);
  const int  wElems   = Dd * Dd;
  const int  wBlocks  = (wElems + eltBlk - 1) / eltBlk;
  const int  gemmBlks = (BN / GT_M) * (Dd / GT_N);           // 64 * 16 = 1024

  // 1) converts + weight transposes
  cvt_f32_bf16<<<muBlocks, eltBlk, 0, stream>>>(mu, mu_bf, totalMu);
  transpose_cvt<<<wBlocks, eltBlk, 0, stream>>>(Wq,  WqT,   Dd, Dd, 0);
  transpose_cvt<<<wBlocks, eltBlk, 0, stream>>>(Wk,  WkT,   Dd, Dd, 0);
  transpose_cvt<<<wBlocks, eltBlk, 0, stream>>>(Wv,  WvT,   Dd, Dd, 0);
  transpose_cvt<<<wBlocks, eltBlk, 0, stream>>>(Wm1, Wm1aT, Dd, Dd, 0);
  transpose_cvt<<<wBlocks, eltBlk, 0, stream>>>(Wm1, Wm1bT, Dd, Dd, Dd);
  transpose_cvt<<<wBlocks, eltBlk, 0, stream>>>(Wm2, Wm2T,  Dd, Dd, 0);
  transpose_cvt<<<wBlocks, eltBlk, 0, stream>>>(Wo,  WoLT,  Dd, Dd, 0);
  transpose_cvt<<<wBlocks, eltBlk, 0, stream>>>(Wo,  WoGT,  Dd, Dd, Dd);

  // 2) projections (bf16 out) + window-MLP partials (f32 out; Ap carries +bm1)
  gemm_bf16<<<gemmBlks, 256, 0, stream>>>(mu_bf, WqT,   bq,      q_bf, BN, Dd, Dd, 1);
  gemm_bf16<<<gemmBlks, 256, 0, stream>>>(mu_bf, WkT,   bk,      k_bf, BN, Dd, Dd, 1);
  gemm_bf16<<<gemmBlks, 256, 0, stream>>>(mu_bf, WvT,   bv,      v_bf, BN, Dd, Dd, 1);
  gemm_bf16<<<gemmBlks, 256, 0, stream>>>(mu_bf, Wm1aT, bm1,     Ap,   BN, Dd, Dd, 0);
  gemm_bf16<<<gemmBlks, 256, 0, stream>>>(mu_bf, Wm1bT, nullptr, Bp,   BN, Dd, Dd, 0);

  // 3) local path: silu-mean then Wm2 (mean commutes with the linear map)
  silu_mean<<<muBlocks, eltBlk, 0, stream>>>(Ap, Bp, s_bf, Nseq, Dd, totalMu);
  gemm_bf16<<<gemmBlks, 256, 0, stream>>>(s_bf, Wm2T, bm2, l_bf, BN, Dd, Dd, 1);

  // 4) causal top-8 attention (WMMA scores in LDS, wave-wide argmax, gather V)
  attn_topk<<<Bb * (Nseq >> 4), 512, (size_t)16 * Nseq * sizeof(float), stream>>>(
      q_bf, k_bf, v_bf, pher, g_bf, Nseq, Dd);

  // 5) output: concat(local, global) @ Wo + bo  ==  local@Wo[:D] + global@Wo[D:]
  gemm_bf16<<<gemmBlks, 256, 0, stream>>>(l_bf, WoLT, bo,      out, BN, Dd, Dd, 0);
  gemm_bf16<<<gemmBlks, 256, 0, stream>>>(g_bf, WoGT, nullptr, out, BN, Dd, Dd, 2);
}
